// TGWPooling_62560493633865
// MI455X (gfx1250) — matrix operand order repaired and checked by reading, not compiled, not run on hardware
//
#include <hip/hip_runtime.h>
#include <hip/hip_bf16.h>

#define N_GRAPHS   256
#define N_NODES    128
#define N_TPLT     16
#define N_TNODES   10
#define EPS        0.1f
#define OUTER_ITERS 5
#define SINKHORN_ITERS 20
#define PSTRIDE    17   // 16 cols + 1 pad (f32 words) -> bank-conflict friendly

typedef __attribute__((ext_vector_type(16))) __bf16 v16bf;
typedef __attribute__((ext_vector_type(8)))  float  v8f;

// ---------------------------------------------------------------------------
// Kernel 1: zero the bit-packed adjacency workspace
// ---------------------------------------------------------------------------
__global__ void zero_bits_kernel(unsigned int* __restrict__ bits, int n) {
    int i = blockIdx.x * blockDim.x + threadIdx.x;
    if (i < n) bits[i] = 0u;
}

// ---------------------------------------------------------------------------
// Kernel 2: scatter edges into bit-packed symmetric adjacency
// bits layout: [graph][row 0..127][4 u32 words] (128 bits per row)
// ---------------------------------------------------------------------------
__global__ void scatter_edges_kernel(const int* __restrict__ ei, int n_edges,
                                     unsigned int* __restrict__ bits) {
    int e = blockIdx.x * blockDim.x + threadIdx.x;
    if (e >= n_edges) return;
    int s = ei[e];              // global src id
    int d = ei[n_edges + e];    // global dst id
    int g = s >> 7;             // batch[src] = src / 128
    int r = s & 127;
    int c = d - (g << 7);       // local dst (same graph by construction)
    atomicOr(&bits[(g * 128 + r) * 4 + (c >> 5)], 1u << (c & 31));
    atomicOr(&bits[(g * 128 + c) * 4 + (r >> 5)], 1u << (r & 31)); // symmetrize
}

// ---------------------------------------------------------------------------
// Kernel 3: one wave32 per (graph b, template t) pair.
// ---------------------------------------------------------------------------
__global__ __launch_bounds__(32) void gw_kernel(
    const unsigned int* __restrict__ bits,  // [256][128][4] u32
    const float* __restrict__ C2,           // [16][10][10]
    const float* __restrict__ q0,           // [16][10]
    float* __restrict__ out)                // [256][16]
{
    __shared__ float sP[N_NODES * PSTRIDE];   // transport plan (row-major, 16-col padded)
    __shared__ float sK[N_NODES * PSTRIDE];   // cross1 = C1 @ P (WMMA output buffer)
    __shared__ uint4 sBits4[N_NODES];         // 128 bits per row, one uint4 per row
    __shared__ uint4 sTab[256];               // byte -> 4 packed bf16-pair words
    __shared__ float sC2[N_TNODES * 11];

    unsigned int* sBits = (unsigned int*)sBits4;

    const int lane = threadIdx.x;
    const int blk  = blockIdx.x;
    const int b    = blk >> 4;
    const int t    = blk & 15;
    const int nCol = lane & 15;
    const int half = lane >> 4;

    // ---- load per-block constants ----
    for (int i = lane; i < N_NODES * 4; i += 32) sBits[i] = bits[b * N_NODES * 4 + i];
    for (int i = lane; i < N_TNODES * N_TNODES; i += 32) {
        int r = i / N_TNODES, c = i % N_TNODES;
        sC2[r * 11 + c] = C2[(t * N_TNODES + r) * N_TNODES + c];
    }
    // ---- build byte -> bf16x2 x4 expansion table (bit -> 1.0bf16) ----
    for (int i = lane; i < 256; i += 32) {
        uint4 e;
        unsigned int w0 = ((i >> 0) & 1) ? 0x3F80u : 0u, w1 = ((i >> 1) & 1) ? 0x3F800000u : 0u;
        e.x = w0 | w1;
        w0 = ((i >> 2) & 1) ? 0x3F80u : 0u; w1 = ((i >> 3) & 1) ? 0x3F800000u : 0u;
        e.y = w0 | w1;
        w0 = ((i >> 4) & 1) ? 0x3F80u : 0u; w1 = ((i >> 5) & 1) ? 0x3F800000u : 0u;
        e.z = w0 | w1;
        w0 = ((i >> 6) & 1) ? 0x3F80u : 0u; w1 = ((i >> 7) & 1) ? 0x3F800000u : 0u;
        e.w = w0 | w1;
        sTab[i] = e;
    }
    __syncthreads();

    // ---- q = softmax(q0[t]) (redundant per lane, 10 elems) ----
    float qv[N_TNODES];
    {
        float mx = -1e30f;
        #pragma unroll
        for (int m = 0; m < N_TNODES; m++) { qv[m] = q0[t * N_TNODES + m]; mx = fmaxf(mx, qv[m]); }
        float s = 0.f;
        #pragma unroll
        for (int m = 0; m < N_TNODES; m++) { qv[m] = __expf(qv[m] - mx); s += qv[m]; }
        float inv = 1.f / s;
        #pragma unroll
        for (int m = 0; m < N_TNODES; m++) qv[m] *= inv;
    }

    // ---- f2q[m] = sum_j C2[m][j]^2 * q[j] ----
    float f2q[N_TNODES];
    #pragma unroll
    for (int m = 0; m < N_TNODES; m++) {
        float acc = 0.f;
        #pragma unroll
        for (int j = 0; j < N_TNODES; j++) { float c = sC2[m * 11 + j]; acc += c * c * qv[j]; }
        f2q[m] = acc;
    }

    // ---- f1p[n] = deg(n)/128 for my 4 rows (n = lane + 32*i) ----
    float f1p[4];
    #pragma unroll
    for (int i = 0; i < 4; i++) {
        uint4 w = sBits4[lane + 32 * i];
        int d = __popc(w.x) + __popc(w.y) + __popc(w.z) + __popc(w.w);
        f1p[i] = (float)d * (1.0f / 128.0f);
    }

    // ---- P0 = p q^T (pad cols 10..15 = 0, stay 0 forever) ----
    #pragma unroll
    for (int i = 0; i < 4; i++) {
        int n = lane + 32 * i;
        #pragma unroll
        for (int m = 0; m < N_TNODES; m++) sP[n * PSTRIDE + m] = (1.0f / 128.0f) * qv[m];
        #pragma unroll
        for (int m = N_TNODES; m < 16; m++) sP[n * PSTRIDE + m] = 0.f;
    }
    __syncthreads();

    // ---- cross1 = C1 @ P via bf16 WMMA; result [128][16] into sK ----
    auto do_cross1 = [&]() {
        union Frag { unsigned int u[8]; v16bf v; };
        // B fragments (one per K-tile, shared across M-tiles).
        // B 32x16 bf16: lane -> col N = lane%16; VGPR r -> rows k = 2r+8*half+(r>=4?8:0), k+1
        Frag bfr[4];
        #pragma unroll
        for (int kt = 0; kt < 4; kt++) {
            #pragma unroll
            for (int r = 0; r < 8; r++) {
                int k = 32 * kt + 2 * r + 8 * half + ((r >= 4) ? 8 : 0);
                unsigned int u0 = __float_as_uint(sP[k * PSTRIDE + nCol]);
                unsigned int u1 = __float_as_uint(sP[(k + 1) * PSTRIDE + nCol]);
                bfr[kt].u[r] = ((u0 + 0x8000u) >> 16) | ((u1 + 0x8000u) & 0xFFFF0000u);
            }
        }
        #pragma unroll
        for (int mt = 0; mt < 8; mt++) {
            // one b128 pulls all 4 bitmap words for this lane's row
            uint4 wrow = sBits4[mt * 16 + (lane & 15)];
            v8f acc = {};
            #pragma unroll
            for (int kt = 0; kt < 4; kt++) {
                unsigned int w = (kt == 0) ? wrow.x : (kt == 1) ? wrow.y : (kt == 2) ? wrow.z : wrow.w;
                // A 16x32 bf16: lane -> row M = lane%16; regs 0..3 use bits
                // [8*half .. 8*half+7], regs 4..7 use bits [16+8*half .. 16+8*half+7]
                unsigned int b0 = (w >> (8 * half)) & 0xFFu;
                unsigned int b1 = (w >> (16 + 8 * half)) & 0xFFu;
                Frag a;
                uint4 lo = sTab[b0];
                uint4 hi = sTab[b1];
                a.u[0] = lo.x; a.u[1] = lo.y; a.u[2] = lo.z; a.u[3] = lo.w;
                a.u[4] = hi.x; a.u[5] = hi.y; a.u[6] = hi.z; a.u[7] = hi.w;
                acc = __builtin_amdgcn_wmma_f32_16x16x32_bf16(
                        false, a.v, false, bfr[kt].v, (short)0, acc, false, false);
            }
            // D layout: VGPR e, lane L -> M = e + 8*(L/16), N = L%16
            #pragma unroll
            for (int e = 0; e < 8; e++) {
                int n = mt * 16 + e + 8 * half;
                sK[n * PSTRIDE + nCol] = acc[e];
            }
        }
    };

    // =========================== outer GW loop ===========================
    for (int it = 0; it < OUTER_ITERS; it++) {
        do_cross1();
        __syncthreads();

        // K = exp(-2*(constC - cross1 @ (2*C2)^T)/EPS), kept in registers (4 rows/lane)
        float kr[4][N_TNODES];
        #pragma unroll
        for (int i = 0; i < 4; i++) {
            int n = lane + 32 * i;
            float row[N_TNODES];
            #pragma unroll
            for (int m = 0; m < N_TNODES; m++) row[m] = sK[n * PSTRIDE + m];
            #pragma unroll
            for (int m = 0; m < N_TNODES; m++) {
                float cross = 0.f;
                #pragma unroll
                for (int l = 0; l < N_TNODES; l++) cross += row[l] * 2.0f * sC2[m * 11 + l];
                float tens = f1p[i] + f2q[m] - cross;
                kr[i][m] = __expf(-(2.0f * tens) * (1.0f / EPS));
            }
        }

        // -------- Sinkhorn (register-resident K): u = p/(Kv), v = q/(K^T u) --------
        float u[4], vv[N_TNODES];
        #pragma unroll
        for (int i = 0; i < 4; i++) u[i] = 1.0f;
        #pragma unroll
        for (int m = 0; m < N_TNODES; m++) vv[m] = 1.0f;

        for (int s = 0; s < SINKHORN_ITERS; s++) {
            #pragma unroll
            for (int i = 0; i < 4; i++) {
                float kv = 0.f;
                #pragma unroll
                for (int m = 0; m < N_TNODES; m++) kv += kr[i][m] * vv[m];
                u[i] = (1.0f / 128.0f) / kv;
            }
            float cs[N_TNODES];
            #pragma unroll
            for (int m = 0; m < N_TNODES; m++) cs[m] = 0.f;
            #pragma unroll
            for (int i = 0; i < 4; i++) {
                #pragma unroll
                for (int m = 0; m < N_TNODES; m++) cs[m] += kr[i][m] * u[i];
            }
            #pragma unroll
            for (int m = 0; m < N_TNODES; m++) {
                float x = cs[m];
                #pragma unroll
                for (int off = 16; off >= 1; off >>= 1) x += __shfl_xor(x, off, 32);
                vv[m] = qv[m] / x;
            }
        }

        // P = diag(u) K diag(v) -> LDS (row-major for the next WMMA's B fragments)
        #pragma unroll
        for (int i = 0; i < 4; i++) {
            int n = lane + 32 * i;
            #pragma unroll
            for (int m = 0; m < N_TNODES; m++)
                sP[n * PSTRIDE + m] = u[i] * kr[i][m] * vv[m];
        }
        __syncthreads();
    }

    // ============ objective: sum( (constC - cross) * P ) ============
    do_cross1();
    __syncthreads();

    float local = 0.f;
    #pragma unroll
    for (int i = 0; i < 4; i++) {
        int n = lane + 32 * i;
        float row[N_TNODES];
        #pragma unroll
        for (int m = 0; m < N_TNODES; m++) row[m] = sK[n * PSTRIDE + m];
        #pragma unroll
        for (int m = 0; m < N_TNODES; m++) {
            float cross = 0.f;
            #pragma unroll
            for (int l = 0; l < N_TNODES; l++) cross += row[l] * 2.0f * sC2[m * 11 + l];
            float tens = f1p[i] + f2q[m] - cross;
            local += tens * sP[n * PSTRIDE + m];
        }
    }
    #pragma unroll
    for (int off = 16; off >= 1; off >>= 1) local += __shfl_xor(local, off, 32);
    if (lane == 0) out[blk] = local;   // out[b*16 + t]
}

// ---------------------------------------------------------------------------
extern "C" void kernel_launch(void* const* d_in, const int* in_sizes, int n_in,
                              void* d_out, int out_size, void* d_ws, size_t ws_size,
                              hipStream_t stream) {
    (void)n_in; (void)out_size; (void)ws_size;
    // setup_inputs order: x, edge_index, batch, tplt_adjacencies, tplt_features, q0
    const int*   edge_index = (const int*)d_in[1];
    const float* tplt_adj   = (const float*)d_in[3];
    const float* q0         = (const float*)d_in[5];
    float* out = (float*)d_out;

    unsigned int* bits = (unsigned int*)d_ws;          // 256*128*4 u32 = 512 KB
    const int nwords  = N_GRAPHS * N_NODES * 4;
    const int n_edges = in_sizes[1] / 2;

    zero_bits_kernel<<<(nwords + 255) / 256, 256, 0, stream>>>(bits, nwords);
    scatter_edges_kernel<<<(n_edges + 255) / 256, 256, 0, stream>>>(edge_index, n_edges, bits);
    gw_kernel<<<N_GRAPHS * N_TPLT, 32, 0, stream>>>(bits, tplt_adj, q0, out);
}
